// AdamsGCN_22574348108378
// MI455X (gfx1250) — compile-verified, matrix operand order checked by reading, not run on hardware
//
#include <hip/hip_runtime.h>
#include <hip/hip_bf16.h>

// ---------------------------------------------------------------------------
// GCN forward for MI455X (gfx1250, wave32, WMMA bf16 16x16x32)
// ---------------------------------------------------------------------------

typedef __attribute__((ext_vector_type(16))) __bf16 v16bf;
typedef __attribute__((ext_vector_type(8)))  float  v8f;

#define FEAT 128
#define HID  256
#define SUBG 16

// round-to-nearest-even f32 -> bf16 (bit pattern)
static __device__ __forceinline__ unsigned short f2bf(float f) {
    unsigned int u = __float_as_uint(f);
    unsigned int r = (u + 0x7FFFu + ((u >> 16) & 1u)) >> 16;
    return (unsigned short)r;
}

static __device__ __forceinline__ unsigned int pack2(float lo, float hi) {
    return (unsigned int)f2bf(lo) | ((unsigned int)f2bf(hi) << 16);
}

// ---------------------------------------------------------------------------
// small utility kernels
// ---------------------------------------------------------------------------
__global__ void fill_kernel(float* __restrict__ p, float v, size_t n) {
    size_t i = (size_t)blockIdx.x * blockDim.x + threadIdx.x;
    size_t stride = (size_t)gridDim.x * blockDim.x;
    for (; i < n; i += stride) p[i] = v;
}

__global__ void cvt_bf16_kernel(const float* __restrict__ src,
                                unsigned short* __restrict__ dst, int n) {
    int i = blockIdx.x * blockDim.x + threadIdx.x;
    if (i < n) dst[i] = f2bf(src[i]);
}

__global__ void degree_kernel(const int* __restrict__ dst_idx,
                              float* __restrict__ deg, int E) {
    int e = blockIdx.x * blockDim.x + threadIdx.x;
    if (e < E) atomicAdd(&deg[dst_idx[e]], 1.0f);
}

__global__ void dinv_kernel(const float* __restrict__ deg,
                            float* __restrict__ dinv, int n) {
    int i = blockIdx.x * blockDim.x + threadIdx.x;
    if (i < n) {
        float d = deg[i];
        dinv[i] = d > 0.0f ? rsqrtf(d) : 0.0f;
    }
}

// per-edge GCN norm, computed once and reused by both layers
__global__ void norm_kernel(const int* __restrict__ src_idx,
                            const int* __restrict__ dst_idx,
                            const float* __restrict__ dinv,
                            float* __restrict__ norm, int E) {
    int e = blockIdx.x * blockDim.x + threadIdx.x;
    if (e < E) norm[e] = dinv[src_idx[e]] * dinv[dst_idx[e]];
}

// ---------------------------------------------------------------------------
// Tiled GEMM: H[nrows x HID] = X[nrows x KDIM] @ Wb[KDIM x HID]
// Block: 256 threads = 8 waves. Block tile: 256 (M) x 32 (N). K-step 32.
// Each wave owns a 32x32 slice -> four 16x16 WMMA accumulators.
// LDS tiles are stored in WMMA *fragment order* (per-lane contiguous 32B),
// so fragment loads are 2x ds_load_b128 instead of 16x ds_load_u16.
// ISA 16-bit A/B layout: lane half hl carries K+8*hl; VGPR v packs K pair
//   {2v, 2v+1} for v<4 and {16+2(v-4), 17+2(v-4)} for v>=4 (low half = even K).
// ---------------------------------------------------------------------------
template <int KDIM>
__global__ __launch_bounds__(256)
void gemm_bf16_kernel(const float* __restrict__ X,
                      const unsigned short* __restrict__ Wb,
                      float* __restrict__ H, int nrows) {
    // A: 16 row-tiles x 32 lanes x 8 uints (fragment-major) = 16 KB
    __shared__ __align__(32) unsigned int aSw[16 * 32 * 8];
    // B: 2 n-tiles x 32 lanes x 8 uints = 2 KB
    __shared__ __align__(32) unsigned int bSw[2 * 32 * 8];

    const int t    = threadIdx.x;
    const int wave = t >> 5;
    const int lane = t & 31;
    const int hl   = lane >> 4;
    const int r    = lane & 15;
    const int rowBase = blockIdx.x * 256;
    const int colBase = blockIdx.y * 32;

    v8f c00 = {}, c01 = {}, c10 = {}, c11 = {};

    for (int k0 = 0; k0 < KDIM; k0 += 32) {
        // ---- stage A: 256 rows x 32 cols fp32 -> bf16, fragment layout ----
        // each thread converts one float4 (4 consecutive K) per iteration
#pragma unroll
        for (int it = 0; it < 8; ++it) {
            int q  = t + it * 256;            // quad id, 0..2047
            int rr = q >> 3;                  // row in tile (8 quads per row)
            int c4 = (q & 7) * 4;             // K offset, multiple of 4
            const float4 xv =
                *(const float4*)&X[(size_t)(rowBase + rr) * KDIM + k0 + c4];
            int shl = (c4 >> 3) & 1;
            int kb  = c4 - shl * 8;
            int vg  = (kb < 8) ? (kb >> 1) : (4 + ((kb - 16) >> 1));  // even
            int ln  = shl * 16 + (rr & 15);
            unsigned int widx = (unsigned)((rr >> 4) * 256 + ln * 8 + vg);
            *(uint2*)&aSw[widx] = make_uint2(pack2(xv.x, xv.y), pack2(xv.z, xv.w));
        }
        // ---- stage B: 32 x 32 bf16 slice, fragment layout (1 iter) ----
        {
            int cc = t & 31;
            int km = (t >> 5) * 4;            // K base, multiple of 4
            size_t wb = (size_t)(k0 + km) * HID + colBase + cc;
            unsigned short w0 = Wb[wb];
            unsigned short w1 = Wb[wb + HID];
            unsigned short w2 = Wb[wb + 2 * HID];
            unsigned short w3 = Wb[wb + 3 * HID];
            int shl = (km >> 3) & 1;
            int kb  = km - shl * 8;
            int vg  = (kb < 8) ? (kb >> 1) : (4 + ((kb - 16) >> 1));
            int ln  = shl * 16 + (cc & 15);
            int nt  = cc >> 4;
            *(uint2*)&bSw[nt * 256 + ln * 8 + vg] =
                make_uint2((unsigned)w0 | ((unsigned)w1 << 16),
                           (unsigned)w2 | ((unsigned)w3 << 16));
        }
        __syncthreads();

        // ---- fragment loads: contiguous 32B per lane -> ds_load_b128 x2 ----
        v16bf aLo = *(const v16bf*)&aSw[(wave * 2 + 0) * 256 + lane * 8];
        v16bf aHi = *(const v16bf*)&aSw[(wave * 2 + 1) * 256 + lane * 8];
        v16bf b0  = *(const v16bf*)&bSw[lane * 8];
        v16bf b1  = *(const v16bf*)&bSw[256 + lane * 8];

        c00 = __builtin_amdgcn_wmma_f32_16x16x32_bf16(false, aLo, false, b0,
                                                      (short)0, c00, false, false);
        c01 = __builtin_amdgcn_wmma_f32_16x16x32_bf16(false, aLo, false, b1,
                                                      (short)0, c01, false, false);
        c10 = __builtin_amdgcn_wmma_f32_16x16x32_bf16(false, aHi, false, b0,
                                                      (short)0, c10, false, false);
        c11 = __builtin_amdgcn_wmma_f32_16x16x32_bf16(false, aHi, false, b1,
                                                      (short)0, c11, false, false);
        __syncthreads();
    }

    // C/D layout: VGPR i -> M = i + 8*hl (within 16x16 tile), N = lane&15
#pragma unroll
    for (int i = 0; i < 8; ++i) {
        int m0 = rowBase + wave * 32 + hl * 8 + i;
        int m1 = m0 + 16;
        H[(size_t)m0 * HID + colBase + r]      = c00[i];
        H[(size_t)m0 * HID + colBase + 16 + r] = c01[i];
        H[(size_t)m1 * HID + colBase + r]      = c10[i];
        H[(size_t)m1 * HID + colBase + 16 + r] = c11[i];
    }
}

// ---------------------------------------------------------------------------
// Edge aggregation: agg[dst] += h[src] * norm[e]
// 64-edge metadata batches staged in LDS (one coalesced load per batch),
// then 256 channel-threads do one 1KB coalesced gather + 256 f32 atomics
// per edge. h (134 MB) is L2-resident on MI455X (192 MB L2).
// ---------------------------------------------------------------------------
#define EBATCH 64
__global__ __launch_bounds__(256)
void agg_kernel(const float* __restrict__ h,
                const int* __restrict__ src_idx,
                const int* __restrict__ dst_idx,
                const float* __restrict__ norm,
                float* __restrict__ agg, int E) {
    __shared__ int   sS[EBATCH];
    __shared__ int   dS[EBATCH];
    __shared__ float nS[EBATCH];
    const int c = threadIdx.x;
    for (int base = blockIdx.x * EBATCH; base < E; base += gridDim.x * EBATCH) {
        int nb = min(EBATCH, E - base);
        if (c < nb) {
            sS[c] = src_idx[base + c];
            dS[c] = dst_idx[base + c];
            nS[c] = norm[base + c];
        }
        __syncthreads();
        for (int j = 0; j < nb; ++j) {
            int   s  = sS[j];
            int   d  = dS[j];
            float nm = nS[j];
            atomicAdd(&agg[(size_t)d * HID + c], h[(size_t)s * HID + c] * nm);
        }
        __syncthreads();
    }
}

// ---------------------------------------------------------------------------
// Post: buf = relu(agg + h_self * dinv^2 + bias)   (self-loop fused here)
// ---------------------------------------------------------------------------
__global__ void post_kernel(float* __restrict__ agg,
                            const float* __restrict__ h,
                            const float* __restrict__ dinv,
                            const float* __restrict__ bias, size_t n) {
    size_t i = (size_t)blockIdx.x * blockDim.x + threadIdx.x;
    size_t stride = (size_t)gridDim.x * blockDim.x;
    for (; i < n; i += stride) {
        int node = (int)(i >> 8);
        float di = dinv[node];
        float v  = agg[i] + h[i] * di * di + bias[i & (HID - 1)];
        agg[i] = fmaxf(v, 0.0f);
    }
}

// ---------------------------------------------------------------------------
// Final: out[g] = sum_{j<16,c<256} relu(agg2+self+b2)[16g+j,c]*Wout[j*256+c]+bout
// ---------------------------------------------------------------------------
__global__ __launch_bounds__(256)
void final_kernel(const float* __restrict__ agg2,
                  const float* __restrict__ h2,
                  const float* __restrict__ dinv,
                  const float* __restrict__ b2,
                  const float* __restrict__ Wout,
                  const float* __restrict__ bout,
                  float* __restrict__ out) {
    __shared__ float red[256];
    const int g = blockIdx.x;
    const int c = threadIdx.x;
    float sum = 0.0f;
#pragma unroll
    for (int j = 0; j < SUBG; ++j) {
        int node = g * SUBG + j;
        float di = dinv[node];
        size_t idx = (size_t)node * HID + c;
        float v = agg2[idx] + h2[idx] * di * di + b2[c];
        v = fmaxf(v, 0.0f);
        sum += v * Wout[j * HID + c];
    }
    red[c] = sum;
    __syncthreads();
#pragma unroll
    for (int s = 128; s > 0; s >>= 1) {
        if (c < s) red[c] += red[c + s];
        __syncthreads();
    }
    if (c == 0) out[g] = red[0] + bout[0];
}

// ---------------------------------------------------------------------------
// launch
// ---------------------------------------------------------------------------
extern "C" void kernel_launch(void* const* d_in, const int* in_sizes, int n_in,
                              void* d_out, int out_size, void* d_ws, size_t ws_size,
                              hipStream_t stream) {
    const float* x    = (const float*)d_in[0];
    const int*   ei   = (const int*)d_in[1];     // [2,E] row-major: src then dst
    const float* W1   = (const float*)d_in[2];
    const float* b1   = (const float*)d_in[3];
    const float* W2   = (const float*)d_in[4];
    const float* b2   = (const float*)d_in[5];
    const float* Wout = (const float*)d_in[6];
    const float* bout = (const float*)d_in[7];
    float* out = (float*)d_out;

    const int N = in_sizes[0] / FEAT;      // 131072
    const int E = in_sizes[1] / 2;         // 4194304
    const int* src_idx = ei;
    const int* dst_idx = ei + E;

    // workspace carve-up
    char* base = (char*)d_ws;
    float* deg  = (float*)base;                     base += (size_t)N * sizeof(float);
    float* dinv = (float*)base;                     base += (size_t)N * sizeof(float);
    float* bufA = (float*)base;                     base += (size_t)N * HID * sizeof(float);
    float* bufB = (float*)base;                     base += (size_t)N * HID * sizeof(float);
    float* nrm  = (float*)base;                     base += (size_t)E * sizeof(float);
    unsigned short* Wb1 = (unsigned short*)base;    base += (size_t)FEAT * HID * sizeof(unsigned short);
    unsigned short* Wb2 = (unsigned short*)base;

    const size_t NH = (size_t)N * HID;

    // 1) init: deg = 1 (self-loops), agg1 buffer = 0
    fill_kernel<<<1024, 256, 0, stream>>>(deg, 1.0f, (size_t)N);
    fill_kernel<<<4096, 256, 0, stream>>>(bufB, 0.0f, NH);

    // 2) convert weights to bf16 (tiny, L2-resident afterwards)
    cvt_bf16_kernel<<<(FEAT * HID + 255) / 256, 256, 0, stream>>>(W1, Wb1, FEAT * HID);
    cvt_bf16_kernel<<<(HID * HID + 255) / 256, 256, 0, stream>>>(W2, Wb2, HID * HID);

    // 3) degrees, dinv, per-edge norms (norms reused by both layers)
    degree_kernel<<<(E + 255) / 256, 256, 0, stream>>>(dst_idx, deg, E);
    dinv_kernel<<<(N + 255) / 256, 256, 0, stream>>>(deg, dinv, N);
    norm_kernel<<<(E + 255) / 256, 256, 0, stream>>>(src_idx, dst_idx, dinv, nrm, E);

    // 4) layer 1: h1 = x @ W1  (WMMA bf16)
    dim3 g1(N / 256, HID / 32);
    gemm_bf16_kernel<FEAT><<<g1, 256, 0, stream>>>(x, Wb1, bufA, N);

    // 5) aggregate layer 1 edges (L2-resident scatter-add)
    agg_kernel<<<8192, 256, 0, stream>>>(bufA, src_idx, dst_idx, nrm, bufB, E);

    // 6) bufB = relu(agg1 + h1*dinv^2 + b1)
    post_kernel<<<4096, 256, 0, stream>>>(bufB, bufA, dinv, b1, NH);

    // 7) layer 2: h2 = bufB @ W2  (WMMA bf16) -> bufA
    gemm_bf16_kernel<HID><<<g1, 256, 0, stream>>>(bufB, Wb2, bufA, N);

    // 8) zero agg2, aggregate layer 2 edges
    fill_kernel<<<4096, 256, 0, stream>>>(bufB, 0.0f, NH);
    agg_kernel<<<8192, 256, 0, stream>>>(bufA, src_idx, dst_idx, nrm, bufB, E);

    // 9) fused relu + reshape + GEMV epilogue
    final_kernel<<<N / SUBG, 256, 0, stream>>>(bufB, bufA, dinv, b2, Wout, bout, out);
}
